// LSTMCell_29360396435535
// MI455X (gfx1250) — matrix operand order repaired and checked by reading, not compiled, not run on hardware
//
#include <hip/hip_runtime.h>
#include <hip/hip_bf16.h>

// ---------------------------------------------------------------------------
// LSTM (N=64, L=512, H=1024) for gfx1250 (MI455X).
//   g = [h, x_t] @ W^T + b  =  h@Wh^T (sequential)  +  x@Wx^T + b (hoisted
//   into one 32768 x 4096 x 1024 bf16 WMMA GEMM over all (n,t) rows).
// Recurrent phase: 256 waves; wave (m,nh) owns a 16x16 (batch x hidden) tile
// for all 512 steps; cell state c stays in VGPRs; h double-buffered bf16.
// The recurrence is independent per batch row -> sync domain is one m-group
// (8 blocks, 64 waves), each with its own atomic step counter.
// All matrix math: v_wmma_f32_16x16x32_bf16.
// ---------------------------------------------------------------------------

typedef __attribute__((ext_vector_type(16))) __bf16 v16bf;
typedef __attribute__((ext_vector_type(8)))  __bf16 v8bf;
typedef __attribute__((ext_vector_type(8)))  float  v8f;

#define NB  64      // batch
#define LT  512     // timesteps
#define HD  1024    // hidden
#define H2  2048    // 2H  (W row length)
#define H4  4096    // 4H  (gate width)

// ---- bf16 helpers (manual RNE; storage is ushort, reinterpret as __bf16) ---
__device__ __forceinline__ unsigned short f2bf(float f) {
    unsigned u = __float_as_uint(f);
    u += 0x7FFFu + ((u >> 16) & 1u);
    return (unsigned short)(u >> 16);
}
__device__ __forceinline__ float bf2f(unsigned short s) {
    return __uint_as_float(((unsigned)s) << 16);
}
__device__ __forceinline__ v16bf cat16(v8bf lo, v8bf hi) {
    return __builtin_shufflevector(lo, hi, 0,1,2,3,4,5,6,7,8,9,10,11,12,13,14,15);
}
__device__ __forceinline__ float rcp_fast(float x) { return __builtin_amdgcn_rcpf(x); }
__device__ __forceinline__ float sigmoidf_fast(float x) {
    return rcp_fast(1.0f + __expf(-x));                 // v_exp + v_rcp
}
__device__ __forceinline__ float tanhf_fast(float x) {
    float e = __expf(-2.0f * fabsf(x));                 // in (0,1], no overflow
    float r = (1.0f - e) * rcp_fast(1.0f + e);
    return copysignf(r, x);
}

// ---- kernel 1: W fp32 -> bf16 (row-major, 4096 x 2048) ---------------------
__global__ void k_conv_w(const float* __restrict__ W, unsigned short* __restrict__ Wbf) {
    const int n = H4 * H2;
    for (int i = blockIdx.x * blockDim.x + threadIdx.x; i < n; i += gridDim.x * blockDim.x)
        Wbf[i] = f2bf(W[i]);
}

// ---- kernel 2: x fp32 (N,L,H) -> bf16 time-major ((l*64+n), H) -------------
__global__ void k_conv_x(const float* __restrict__ x, unsigned short* __restrict__ xbf) {
    const int total = NB * LT * HD;
    for (int d = blockIdx.x * blockDim.x + threadIdx.x; d < total; d += gridDim.x * blockDim.x) {
        int k = d & (HD - 1);
        int r = d >> 10;          // l*64 + n
        int n = r & 63;
        int l = r >> 6;
        xbf[d] = f2bf(x[(n * LT + l) * HD + k]);
    }
}

// ---- kernel 3: zero per-m barrier counters + both h buffers ----------------
__global__ void k_init(unsigned* __restrict__ bar,
                       unsigned short* __restrict__ h0,
                       unsigned short* __restrict__ h1) {
    int i = blockIdx.x * blockDim.x + threadIdx.x;
    if (i < 4) bar[i * 64] = 0u;               // one counter per m-group, own cacheline
    for (int d = i; d < NB * HD; d += gridDim.x * blockDim.x) { h0[d] = 0; h1[d] = 0; }
}

// ---- kernel 4: Gpre = x @ Wx^T + b ----------------------------------------
// 2x2 register tile per wave (32x32 output): 8 b128 loads -> 4 WMMAs,
// halving VMEM issue pressure vs 1 tile/wave (load pipe ~1/cyc vs WMMA ~4cyc).
__global__ void __launch_bounds__(256) k_gemm_x(const unsigned short* __restrict__ xbf,
                                                const unsigned short* __restrict__ Wbf,
                                                const float* __restrict__ bias,
                                                unsigned short* __restrict__ Gpre) {
    const int lane  = threadIdx.x & 31;
    const int wave  = threadIdx.x >> 5;
    const int gw    = blockIdx.x * (blockDim.x >> 5) + wave;
    const int nwv   = gridDim.x * (blockDim.x >> 5);     // multiple of 128 by launch config
    const int lm    = lane & 15;
    const int hi    = lane >> 4;
    const int ah    = hi ? 8 : 0;    // A half-wave K offset
    const int bh    = hi ? 8 : 0;    // B half-wave K offset
    const int MT2   = NB * LT / 32;  // 1024 row-pair tiles
    const int NT2   = H4 / 32;       // 128 col-pair tiles
    const int NTIL  = MT2 * NT2;

    for (int unit = gw; unit < NTIL; unit += nwv) {
        int nt2 = unit & (NT2 - 1);  // constant per wave (nwv % 128 == 0) -> B stays cache-hot
        int mt2 = unit >> 7;
        const int rowb = mt2 * 32;
        const int colb = nt2 * 32;
        const unsigned short* ap0 = xbf + (size_t)(rowb + lm) * HD;
        const unsigned short* ap1 = ap0 + (size_t)16 * HD;
        const unsigned short* bp0 = Wbf + (size_t)(colb + lm) * H2 + HD;  // x-part of W
        const unsigned short* bp1 = bp0 + (size_t)16 * H2;
        v8f a00 = {}, a01 = {}, a10 = {}, a11 = {};
        #pragma unroll 2
        for (int k0 = 0; k0 < HD; k0 += 32) {
            v16bf A0 = cat16(*(const v8bf*)(ap0 + k0 + ah), *(const v8bf*)(ap0 + k0 + ah + 16));
            v16bf A1 = cat16(*(const v8bf*)(ap1 + k0 + ah), *(const v8bf*)(ap1 + k0 + ah + 16));
            v16bf B0 = cat16(*(const v8bf*)(bp0 + k0 + bh), *(const v8bf*)(bp0 + k0 + bh + 16));
            v16bf B1 = cat16(*(const v8bf*)(bp1 + k0 + bh), *(const v8bf*)(bp1 + k0 + bh + 16));
            a00 = __builtin_amdgcn_wmma_f32_16x16x32_bf16(false, A0, false, B0, (short)0, a00, false, false);
            a01 = __builtin_amdgcn_wmma_f32_16x16x32_bf16(false, A0, false, B1, (short)0, a01, false, false);
            a10 = __builtin_amdgcn_wmma_f32_16x16x32_bf16(false, A1, false, B0, (short)0, a10, false, false);
            a11 = __builtin_amdgcn_wmma_f32_16x16x32_bf16(false, A1, false, B1, (short)0, a11, false, false);
        }
        const float bv0 = bias[colb + lm];
        const float bv1 = bias[colb + 16 + lm];
        const int hi8 = hi ? 8 : 0;
        unsigned short* cp0 = Gpre + (size_t)(rowb + hi8) * H4 + colb + lm;       // (r0,c0)
        unsigned short* cp2 = Gpre + (size_t)(rowb + 16 + hi8) * H4 + colb + lm;  // (r1,c0)
        #pragma unroll
        for (int v = 0; v < 8; ++v) {
            const size_t ro = (size_t)v * H4;
            cp0[ro]      = f2bf(a00[v] + bv0);
            cp0[ro + 16] = f2bf(a01[v] + bv1);
            cp2[ro]      = f2bf(a10[v] + bv0);
            cp2[ro + 16] = f2bf(a11[v] + bv1);
        }
    }
}

// ---- kernel 5: persistent recurrent LSTM ----------------------------------
// 32 blocks x 256 thr = 256 waves. Block b handles m = b>>3 (batch rows
// m*16..m*16+15); the 8 blocks of an m-group sync via their own counter.
#define GRP_BLK 8
__global__ void __launch_bounds__(256, 1) k_lstm(const unsigned short* __restrict__ Wbf,
                                                 const unsigned short* __restrict__ Gpre,
                                                 unsigned short* __restrict__ h0,
                                                 unsigned short* __restrict__ h1,
                                                 float* __restrict__ out,
                                                 unsigned* __restrict__ bar) {
    const int lane = threadIdx.x & 31;
    const int wave = threadIdx.x >> 5;
    const int m    = blockIdx.x >> 3;               // batch tile 0..3
    const int nh   = (blockIdx.x & 7) * 8 + wave;   // hidden tile 0..63
    unsigned* mybar = bar + m * 64;                 // per-m counter, own cacheline
    const int lm   = lane & 15;
    const int hi   = lane >> 4;
    const int ah   = hi ? 8 : 0;
    const int bh   = hi ? 8 : 0;
    const int rowA = m * 16 + lm;                   // batch row this lane loads for A
    const int col  = nh * 16 + lm;                  // hidden column this lane owns
    const int row0 = m * 16 + (hi ? 8 : 0);         // first C-row this lane owns

    // B base pointers: column j of B == row (g*H + col) of row-major bf16 W
    const unsigned short* bpf = Wbf + (size_t)(0 * HD + col) * H2;
    const unsigned short* bpi = Wbf + (size_t)(1 * HD + col) * H2;
    const unsigned short* bpo = Wbf + (size_t)(2 * HD + col) * H2;
    const unsigned short* bpc = Wbf + (size_t)(3 * HD + col) * H2;

    v8f c = {};   // cell state tile: lives in VGPRs for all 512 steps

    for (int t = 0; t < LT; ++t) {
        const unsigned short* hp = (t & 1) ? h1 : h0;   // read h(t-1)
        unsigned short*       hn = (t & 1) ? h0 : h1;   // write h(t)
        const unsigned short* ap = hp + (size_t)rowA * HD;

        v8f af = {}, ai = {}, ao = {}, ac = {};
        for (int k0 = 0; k0 < HD; k0 += 32) {
            v16bf a = cat16(*(const v8bf*)(ap + k0 + ah), *(const v8bf*)(ap + k0 + ah + 16));
            v16bf b;
            b  = cat16(*(const v8bf*)(bpf + k0 + bh), *(const v8bf*)(bpf + k0 + bh + 16));
            af = __builtin_amdgcn_wmma_f32_16x16x32_bf16(false, a, false, b, (short)0, af, false, false);
            b  = cat16(*(const v8bf*)(bpi + k0 + bh), *(const v8bf*)(bpi + k0 + bh + 16));
            ai = __builtin_amdgcn_wmma_f32_16x16x32_bf16(false, a, false, b, (short)0, ai, false, false);
            b  = cat16(*(const v8bf*)(bpo + k0 + bh), *(const v8bf*)(bpo + k0 + bh + 16));
            ao = __builtin_amdgcn_wmma_f32_16x16x32_bf16(false, a, false, b, (short)0, ao, false, false);
            b  = cat16(*(const v8bf*)(bpc + k0 + bh), *(const v8bf*)(bpc + k0 + bh + 16));
            ac = __builtin_amdgcn_wmma_f32_16x16x32_bf16(false, a, false, b, (short)0, ac, false, false);
        }

        // Gpre tile for (t, rows row0..row0+7, this gate column)
        const unsigned short* gp = Gpre + ((size_t)t * NB + row0) * H4 + col;
        #pragma unroll
        for (int v = 0; v < 8; ++v) {
            const size_t ro = (size_t)v * H4;
            float gf = af[v] + bf2f(gp[ro + 0 * HD]);
            float gi = ai[v] + bf2f(gp[ro + 1 * HD]);
            float go = ao[v] + bf2f(gp[ro + 2 * HD]);
            float gc = ac[v] + bf2f(gp[ro + 3 * HD]);
            float f  = sigmoidf_fast(gf);
            float ig = sigmoidf_fast(gi);
            float o  = sigmoidf_fast(go);
            float cd = tanhf_fast(gc);
            float cn = f * c[v] + ig * cd;
            c[v] = cn;
            float hv = o * tanhf_fast(cn);
            int row = row0 + v;
            hn[(size_t)row * HD + col] = f2bf(hv);
            out[((size_t)row * LT + t) * HD + col] = hv;
        }

        // per-m-group barrier: monotonic counter (zeroed by k_init each call)
        __threadfence();
        __syncthreads();
        if (threadIdx.x == 0) {
            __hip_atomic_fetch_add(mybar, 1u, __ATOMIC_RELEASE, __HIP_MEMORY_SCOPE_AGENT);
            const unsigned target = (unsigned)(t + 1) * GRP_BLK;
            while (__hip_atomic_load(mybar, __ATOMIC_ACQUIRE, __HIP_MEMORY_SCOPE_AGENT) < target)
                __builtin_amdgcn_s_sleep(1);
        }
        __syncthreads();
    }
}

// ---------------------------------------------------------------------------
extern "C" void kernel_launch(void* const* d_in, const int* in_sizes, int n_in,
                              void* d_out, int out_size, void* d_ws, size_t ws_size,
                              hipStream_t stream) {
    const float* x = (const float*)d_in[0];   // (64, 512, 1024) fp32
    const float* W = (const float*)d_in[1];   // (4096, 2048)   fp32
    const float* b = (const float*)d_in[2];   // (4096,)        fp32
    float* out = (float*)d_out;               // (64, 512, 1024) fp32
    char* ws = (char*)d_ws;

    // workspace layout (all offsets 256-aligned)
    size_t off = 0;
    unsigned* bar = (unsigned*)(ws + off);                  off += 1024;                  // 4 counters
    unsigned short* h0   = (unsigned short*)(ws + off);     off += (size_t)NB * HD * 2;   // 128 KB
    unsigned short* h1   = (unsigned short*)(ws + off);     off += (size_t)NB * HD * 2;   // 128 KB
    unsigned short* Wbf  = (unsigned short*)(ws + off);     off += (size_t)H4 * H2 * 2;   // 16 MB
    unsigned short* xbf  = (unsigned short*)(ws + off);     off += (size_t)NB * LT * HD * 2; // 64 MB
    unsigned short* Gpre = (unsigned short*)(ws + off);     off += (size_t)NB * LT * H4 * 2; // 256 MB
    if (ws_size < off) return;   // ~337 MB required

    k_conv_w<<<2048, 256, 0, stream>>>(W, Wbf);
    k_conv_x<<<4096, 256, 0, stream>>>(x, xbf);
    k_init  <<<256,  256, 0, stream>>>(bar, h0, h1);
    k_gemm_x<<<512,  256, 0, stream>>>(xbf, Wbf, b, Gpre);   // nwv = 4096 (mult. of 128)
    k_lstm  <<<32,   256, 0, stream>>>(Wbf, Gpre, h0, h1, out, bar);
}
